// Qwen3MoETransformerBlock_46102178955345
// MI455X (gfx1250) — compile-verified, matrix operand order checked by reading, not run on hardware
//
#include <hip/hip_runtime.h>

// ---------------------------------------------------------------------------
// Qwen3-MoE transformer block for gfx1250 (MI455X), wave32 + WMMA bf16 + TDM.
// B=2, S=1024, D=1024, H=16, KV=4, HD=64, E=16, K(top)=2, INTER=1408.
// All matmuls use v_wmma_f32_16x16x32_bf16 with f32 accumulation.
// A tiles staged into LDS via the Tensor Data Mover (tensor_load_to_lds),
// f32 weights converted to bf16 on-the-fly while staging into LDS.
// ---------------------------------------------------------------------------

typedef __bf16 bf16;
typedef __attribute__((ext_vector_type(16))) __bf16 v16bf;
typedef __attribute__((ext_vector_type(8)))  __bf16 v8bf;
typedef __attribute__((ext_vector_type(8)))  float  v8f;

#define TTOK   2048      // B*S
#define DMODEL 1024
#define NHEAD  16
#define NKV    4
#define HD     64
#define NEXP   16
#define INTER  1408

__device__ inline v16bf make_frag(const bf16* p0, const bf16* p1) {
  v8bf a = *(const v8bf*)p0;
  v8bf b = *(const v8bf*)p1;
  return __builtin_shufflevector(a, b, 0,1,2,3,4,5,6,7,8,9,10,11,12,13,14,15);
}

__device__ inline v8f wmma_bf16(v16bf a, v16bf b, v8f c) {
  return __builtin_amdgcn_wmma_f32_16x16x32_bf16(false, a, false, b, (short)0, c, false, false);
}

// ---------------------------------------------------------------------------
// Tensor Data Mover staging (CDNA5). 2D tile of 2-byte elements -> LDS,
// packed row-major. Issued by one wave; completion via TENSORcnt.
// ---------------------------------------------------------------------------
#if __has_builtin(__builtin_amdgcn_tensor_load_to_lds) && \
    __has_builtin(__builtin_amdgcn_s_wait_tensorcnt)
#define USE_TDM 1
typedef unsigned int u32x4 __attribute__((ext_vector_type(4)));
typedef int          i32x4 __attribute__((ext_vector_type(4)));
typedef int          i32x8 __attribute__((ext_vector_type(8)));
typedef __attribute__((address_space(3))) bf16 lds_bf16_t;

__device__ inline void tdm_load_2d_b16(bf16* lds_dst, const bf16* gsrc,
                                       int tile_cols, int tile_rows, int row_stride) {
  unsigned lds_off = (unsigned)(size_t)(lds_bf16_t*)lds_dst;
  unsigned long long ga = (unsigned long long)(size_t)gsrc;
  u32x4 g0;
  g0[0] = 1u;                                                  // count=1 (valid D#)
  g0[1] = lds_off;                                             // LDS byte address
  g0[2] = (unsigned)(ga & 0xffffffffull);                      // global addr [31:0]
  g0[3] = (unsigned)((ga >> 32) & 0x1ffffffull) | (2u << 30);  // addr[56:32], type=2
  i32x8 g1;
  g1[0] = 1 << 16;                                             // data_size = 2 bytes
  g1[1] = (tile_cols & 0xffff) << 16;                          // tensor_dim0 lo16
  g1[2] = ((tile_cols >> 16) & 0xffff) | ((tile_rows & 0xffff) << 16); // dim0 hi / dim1 lo
  g1[3] = ((tile_rows >> 16) & 0xffff) | ((tile_cols & 0xffff) << 16); // dim1 hi / tile_dim0
  g1[4] = tile_rows & 0xffff;                                  // tile_dim1 (tile_dim2=0)
  g1[5] = row_stride;                                          // tensor_dim0_stride lo32
  g1[6] = 0;                                                   // stride hi / dim1_stride
  g1[7] = 0;
  i32x4 gz = {};
#if defined(__clang_major__) && (__clang_major__ >= 23)
  i32x8 gz8 = {};
  __builtin_amdgcn_tensor_load_to_lds(g0, g1, gz, gz, gz8, 0);
#else
  __builtin_amdgcn_tensor_load_to_lds(g0, g1, gz, gz, 0);
#endif
}
#define TDM_WAIT() __builtin_amdgcn_s_wait_tensorcnt(0)
#endif

__device__ inline float block_reduce_sum256(float v, float* scratch) {
  int tid = threadIdx.x;
  __syncthreads();
  #pragma unroll
  for (int off = 16; off > 0; off >>= 1) v += __shfl_down(v, off);
  if ((tid & 31) == 0) scratch[tid >> 5] = v;
  __syncthreads();
  if (tid == 0) {
    float s = 0.f;
    #pragma unroll
    for (int i = 0; i < 8; i++) s += scratch[i];
    scratch[0] = s;
  }
  __syncthreads();
  return scratch[0];
}

// ---------------------------------------------------------------------------
// Small utility kernels
// ---------------------------------------------------------------------------
__global__ void zero_counts_kernel(int* c) {
  if (threadIdx.x < NEXP) c[threadIdx.x] = 0;
}

__global__ __launch_bounds__(256) void rmsnorm1_kernel(
    const float* __restrict__ x, const float* __restrict__ w, bf16* __restrict__ out) {
  int t = blockIdx.x, tid = threadIdx.x;
  __shared__ float scratch[8];
  float s = 0.f;
  for (int d = tid; d < DMODEL; d += 256) { float v = x[t * DMODEL + d]; s += v * v; }
  s = block_reduce_sum256(s, scratch);
  float r = rsqrtf(s / (float)DMODEL + 1e-6f);
  for (int d = tid; d < DMODEL; d += 256)
    out[t * DMODEL + d] = (bf16)(x[t * DMODEL + d] * r * w[d]);
}

__global__ __launch_bounds__(256) void rmsnorm2_kernel(
    const float* __restrict__ x2, const float* __restrict__ w,
    float* __restrict__ tf, bf16* __restrict__ tb) {
  int t = blockIdx.x, tid = threadIdx.x;
  __shared__ float scratch[8];
  float s = 0.f;
  for (int d = tid; d < DMODEL; d += 256) { float v = x2[t * DMODEL + d]; s += v * v; }
  s = block_reduce_sum256(s, scratch);
  float r = rsqrtf(s / (float)DMODEL + 1e-6f);
  for (int d = tid; d < DMODEL; d += 256) {
    float v = x2[t * DMODEL + d] * r * w[d];
    tf[t * DMODEL + d] = v;
    tb[t * DMODEL + d] = (bf16)v;
  }
}

__global__ void add_residual_kernel(const float* __restrict__ x,
                                    const float* __restrict__ y,
                                    float* __restrict__ out, int n) {
  int i = blockIdx.x * blockDim.x + threadIdx.x;
  if (i < n) out[i] = x[i] + y[i];
}

// ---------------------------------------------------------------------------
// Shared staging helpers for the GEMM kernels.
//   A tile : 64 x 32 bf16, packed row-major in LDS (TDM when available).
//   B tile : 32 x 128 f32 -> bf16, stored transposed [col][k] stride 40,
//            two adjacent k-rows converted + packed per 32-bit ds store.
// ---------------------------------------------------------------------------
__device__ inline void stage_a_tile(bf16* As, const bf16* A, int m0, int k0, int K,
                                    int tid) {
#if defined(USE_TDM)
  if (tid < 32) tdm_load_2d_b16(As, &A[(size_t)m0 * K + k0], 32, 64, K);
#else
  int r = tid >> 2, cg = (tid & 3) << 3;
  *(uint4*)&As[r * 32 + cg] = *(const uint4*)&A[(size_t)(m0 + r) * K + k0 + cg];
#endif
}

__device__ inline void stage_a_wait(int tid) {
#if defined(USE_TDM)
  if (tid < 32) TDM_WAIT();
#endif
}

__device__ inline void stage_b_tile(bf16* Bt, const float* B, int k0, int n0, int N,
                                    int tid) {
  int kp = tid & 15;        // k-pair: kk = 2*kp, 2*kp+1
  int cg = tid >> 4;        // column group: cols cg*8 .. cg*8+7
  const float* s0p = &B[(size_t)(k0 + 2 * kp) * N + n0 + cg * 8];
  const float* s1p = s0p + N;
  #pragma unroll
  for (int i = 0; i < 8; i++) {
    unsigned short lo = __builtin_bit_cast(unsigned short, (bf16)s0p[i]);
    unsigned short hi = __builtin_bit_cast(unsigned short, (bf16)s1p[i]);
    *(unsigned*)&Bt[(cg * 8 + i) * 40 + 2 * kp] = (unsigned)lo | ((unsigned)hi << 16);
  }
}

// ---------------------------------------------------------------------------
// Generic WMMA GEMM: C[M,N](f32) = A[M,K](bf16, row major) * B[K,N](f32 -> bf16)
// Block tile 64x128, K-step 32, 256 threads (8 waves), wave tile 16x64.
// ---------------------------------------------------------------------------
__global__ __launch_bounds__(256) void gemm_bf16_kernel(
    const bf16* __restrict__ A, const float* __restrict__ B, float* __restrict__ C,
    int M, int N, int K) {
  __shared__ __align__(16) bf16 As[64 * 32];
  __shared__ __align__(16) bf16 Bt[128 * 40];
  int m0 = blockIdx.x * 64, n0 = blockIdx.y * 128;
  int tid = threadIdx.x;
  int wid = tid >> 5, lid = tid & 31, l16 = lid & 15, hf = lid >> 4;
  int wm = wid & 3, wn = wid >> 2;
  v8f acc[4] = {};
  for (int k0 = 0; k0 < K; k0 += 32) {
    stage_a_tile(As, A, m0, k0, K, tid);
    stage_b_tile(Bt, B, k0, n0, N, tid);
    stage_a_wait(tid);
    __syncthreads();
    const bf16* ar = &As[(wm * 16 + l16) * 32];
    v16bf afrag = make_frag(ar + 8 * hf, ar + 16 + 8 * hf);
    #pragma unroll
    for (int nt = 0; nt < 4; nt++) {
      const bf16* br = &Bt[(wn * 64 + nt * 16 + l16) * 40];
      v16bf bfrag = make_frag(br + 8 * hf, br + 16 + 8 * hf);
      acc[nt] = wmma_bf16(afrag, bfrag, acc[nt]);
    }
    __syncthreads();
  }
  #pragma unroll
  for (int nt = 0; nt < 4; nt++)
    #pragma unroll
    for (int r = 0; r < 8; r++) {
      int row = m0 + wm * 16 + r + 8 * hf;
      int col = n0 + wn * 64 + nt * 16 + l16;
      C[(size_t)row * N + col] = acc[nt][r];
    }
}

// ---------------------------------------------------------------------------
// q/k per-head RMSNorm + RoPE (+1/sqrt(HD) folded into q), v pass-through.
// One 64-thread block per (token, unit); unit: 0..15 q-heads, 16..19 k, 20..23 v.
// Outputs in [b, head, s, hd] bf16 layout for the attention kernel.
// ---------------------------------------------------------------------------
__global__ __launch_bounds__(64) void qk_rope_kernel(
    const float* __restrict__ qraw, const float* __restrict__ kraw,
    const float* __restrict__ vraw, const float* __restrict__ cosb,
    const float* __restrict__ sinb, const float* __restrict__ qnw,
    const float* __restrict__ knw,
    bf16* __restrict__ qbf, bf16* __restrict__ kbf, bf16* __restrict__ vbf) {
  int blk = blockIdx.x;
  int tok = blk / 24, unit = blk % 24;
  int b = tok >> 10, s = tok & 1023;
  int d = threadIdx.x;
  __shared__ float sh[64];
  __shared__ float ssum[2];
  if (unit < 16) {
    int h = unit;
    float v = qraw[(size_t)tok * (NHEAD * HD) + h * HD + d];
    float p = v * v;
    #pragma unroll
    for (int off = 16; off > 0; off >>= 1) p += __shfl_down(p, off);
    if ((d & 31) == 0) ssum[d >> 5] = p;
    __syncthreads();
    float rms = rsqrtf((ssum[0] + ssum[1]) / (float)HD + 1e-6f);
    float nv = v * rms * qnw[d];
    sh[d] = nv;
    __syncthreads();
    float rot = (d < 32) ? -sh[d + 32] : sh[d - 32];
    float out = (nv * cosb[s * HD + d] + rot * sinb[s * HD + d]) * 0.125f; // /sqrt(64)
    qbf[((size_t)(b * NHEAD + h) * 1024 + s) * HD + d] = (bf16)out;
  } else if (unit < 20) {
    int kh = unit - 16;
    float v = kraw[(size_t)tok * (NKV * HD) + kh * HD + d];
    float p = v * v;
    #pragma unroll
    for (int off = 16; off > 0; off >>= 1) p += __shfl_down(p, off);
    if ((d & 31) == 0) ssum[d >> 5] = p;
    __syncthreads();
    float rms = rsqrtf((ssum[0] + ssum[1]) / (float)HD + 1e-6f);
    float nv = v * rms * knw[d];
    sh[d] = nv;
    __syncthreads();
    float rot = (d < 32) ? -sh[d + 32] : sh[d - 32];
    float out = nv * cosb[s * HD + d] + rot * sinb[s * HD + d];
    kbf[((size_t)(b * NKV + kh) * 1024 + s) * HD + d] = (bf16)out;
  } else {
    int vh = unit - 20;
    vbf[((size_t)(b * NKV + vh) * 1024 + s) * HD + d] =
        (bf16)vraw[(size_t)tok * (NKV * HD) + vh * HD + d];
  }
}

// ---------------------------------------------------------------------------
// Flash-attention with causal mask, GQA (REP=4). One block per (b*H, 64 q rows).
// 128 threads = 4 waves, each wave owns 16 query rows. Key blocks of 32.
// ---------------------------------------------------------------------------
__global__ __launch_bounds__(128) void attn_kernel(
    const bf16* __restrict__ Q, const bf16* __restrict__ Kp,
    const bf16* __restrict__ Vp, bf16* __restrict__ O) {
  int bh = blockIdx.x;
  int b = bh / NHEAD, h = bh % NHEAD;
  int kvh = h >> 2;
  int qb = blockIdx.y * 64;
  const bf16* qbase = Q + (size_t)bh * 1024 * HD;
  const bf16* kbase = Kp + (size_t)(b * NKV + kvh) * 1024 * HD;
  const bf16* vbase = Vp + (size_t)(b * NKV + kvh) * 1024 * HD;

  __shared__ __align__(16) bf16 Kt[32 * 72];      // [key][hd], padded
  __shared__ __align__(16) bf16 Vt[64 * 40];      // [hd][key], transposed, padded
  __shared__ __align__(16) bf16 Ps[4 * 16 * 40];  // per-wave P staging

  int tid = threadIdx.x, wid = tid >> 5, lid = tid & 31, l16 = lid & 15, hf = lid >> 4;
  int qrow0 = qb + wid * 16;

  const bf16* qr = qbase + (size_t)(qrow0 + l16) * HD;
  v16bf qf0 = make_frag(qr + 8 * hf, qr + 16 + 8 * hf);
  v16bf qf1 = make_frag(qr + 32 + 8 * hf, qr + 48 + 8 * hf);

  v8f o[4] = {};
  float m[8], l[8];
  #pragma unroll
  for (int r = 0; r < 8; r++) { m[r] = -1e30f; l[r] = 0.f; }

  int kbmax = (qb + 63) >> 5;
  for (int kb = 0; kb <= kbmax; kb++) {
    { // stage K (row major) and V (transposed)
      int key = tid >> 2, seg = tid & 3;
      const uint4* ksp = (const uint4*)(kbase + (size_t)(kb * 32 + key) * HD + seg * 16);
      *(uint4*)&Kt[key * 72 + seg * 16] = ksp[0];
      *(uint4*)&Kt[key * 72 + seg * 16 + 8] = ksp[1];
      union { uint4 u[2]; bf16 hh[16]; } tv;
      const uint4* vsp = (const uint4*)(vbase + (size_t)(kb * 32 + key) * HD + seg * 16);
      tv.u[0] = vsp[0]; tv.u[1] = vsp[1];
      #pragma unroll
      for (int i = 0; i < 16; i++) Vt[(seg * 16 + i) * 40 + key] = tv.hh[i];
    }
    __syncthreads();

    if (kb * 32 <= qrow0 + 15) {   // wave has at least one unmasked row
      // scores: S = Q * K^T, two 16x16 key tiles, 2 chained WMMAs each (HD=64)
      v8f s0 = {}, s1 = {};
      {
        const bf16* kr = &Kt[l16 * 72];
        s0 = wmma_bf16(qf0, make_frag(kr + 8 * hf, kr + 16 + 8 * hf), s0);
        s0 = wmma_bf16(qf1, make_frag(kr + 32 + 8 * hf, kr + 48 + 8 * hf), s0);
      }
      {
        const bf16* kr = &Kt[(16 + l16) * 72];
        s1 = wmma_bf16(qf0, make_frag(kr + 8 * hf, kr + 16 + 8 * hf), s1);
        s1 = wmma_bf16(qf1, make_frag(kr + 32 + 8 * hf, kr + 48 + 8 * hf), s1);
      }
      // causal mask + online softmax
      float mn[8];
      #pragma unroll
      for (int r = 0; r < 8; r++) {
        int q = qrow0 + r + 8 * hf;
        float a0 = ((kb * 32 + l16) <= q) ? s0[r] : -1e30f;
        float a1 = ((kb * 32 + 16 + l16) <= q) ? s1[r] : -1e30f;
        s0[r] = a0; s1[r] = a1;
        float v = fmaxf(a0, a1);
        v = fmaxf(v, __shfl_xor(v, 1));
        v = fmaxf(v, __shfl_xor(v, 2));
        v = fmaxf(v, __shfl_xor(v, 4));
        v = fmaxf(v, __shfl_xor(v, 8));
        mn[r] = fmaxf(m[r], v);
      }
      #pragma unroll
      for (int r = 0; r < 8; r++) {
        float sc = __expf(m[r] - mn[r]);
        m[r] = mn[r];
        float p0 = __expf(s0[r] - mn[r]);
        float p1 = __expf(s1[r] - mn[r]);
        s0[r] = p0; s1[r] = p1;
        float rs = p0 + p1;
        rs += __shfl_xor(rs, 1);
        rs += __shfl_xor(rs, 2);
        rs += __shfl_xor(rs, 4);
        rs += __shfl_xor(rs, 8);
        l[r] = l[r] * sc + rs;
        #pragma unroll
        for (int nt = 0; nt < 4; nt++) o[nt][r] *= sc;
      }
      // P (C layout) -> LDS -> A layout, then O += P * V
      bf16* pw = &Ps[wid * 16 * 40];
      #pragma unroll
      for (int r = 0; r < 8; r++) {
        pw[(r + 8 * hf) * 40 + l16] = (bf16)s0[r];
        pw[(r + 8 * hf) * 40 + 16 + l16] = (bf16)s1[r];
      }
      const bf16* pr = pw + l16 * 40;
      v16bf pf = make_frag(pr + 8 * hf, pr + 16 + 8 * hf);
      #pragma unroll
      for (int nt = 0; nt < 4; nt++) {
        const bf16* vr = &Vt[(nt * 16 + l16) * 40];
        o[nt] = wmma_bf16(pf, make_frag(vr + 8 * hf, vr + 16 + 8 * hf), o[nt]);
      }
    }
    __syncthreads();
  }
  // epilogue: O / l, write [tok, h*64+hd] bf16
  #pragma unroll
  for (int nt = 0; nt < 4; nt++)
    #pragma unroll
    for (int r = 0; r < 8; r++) {
      int q = qrow0 + r + 8 * hf;
      int col = h * HD + nt * 16 + l16;
      O[(size_t)(b * 1024 + q) * DMODEL + col] = (bf16)(o[nt][r] / l[r]);
    }
}

// ---------------------------------------------------------------------------
// Router: logits = t @ router_w, softmax(16), top-2, per-expert token lists.
// ---------------------------------------------------------------------------
__global__ __launch_bounds__(256) void router_kernel(
    const float* __restrict__ t, const float* __restrict__ rw,
    float* __restrict__ probs, int* __restrict__ counts,
    int* __restrict__ etok, float* __restrict__ ewt) {
  int tok = blockIdx.x, tid = threadIdx.x;
  float acc[NEXP];
  #pragma unroll
  for (int e = 0; e < NEXP; e++) acc[e] = 0.f;
  for (int d = tid; d < DMODEL; d += 256) {
    float tv = t[(size_t)tok * DMODEL + d];
    const float* r = &rw[d * NEXP];
    #pragma unroll
    for (int e = 0; e < NEXP; e++) acc[e] += tv * r[e];
  }
  __shared__ float scratch[8];
  __shared__ float logits[NEXP];
  for (int e = 0; e < NEXP; e++) {
    float s = block_reduce_sum256(acc[e], scratch);
    if (tid == 0) logits[e] = s;
  }
  __syncthreads();
  if (tid == 0) {
    float mx = -1e30f;
    for (int e = 0; e < NEXP; e++) mx = fmaxf(mx, logits[e]);
    float p[NEXP]; float se = 0.f;
    for (int e = 0; e < NEXP; e++) { p[e] = __expf(logits[e] - mx); se += p[e]; }
    int i0 = 0, i1 = -1; float v0 = -1.f, v1 = -1.f;
    for (int e = 0; e < NEXP; e++) {
      p[e] /= se;
      probs[tok * NEXP + e] = p[e];
      if (p[e] > v0) { v1 = v0; i1 = i0; v0 = p[e]; i0 = e; }
      else if (p[e] > v1) { v1 = p[e]; i1 = e; }
    }
    float wn = v0 + v1;
    int p0 = atomicAdd(&counts[i0], 1);
    etok[i0 * TTOK + p0] = tok; ewt[i0 * TTOK + p0] = v0 / wn;
    int p1 = atomicAdd(&counts[i1], 1);
    etok[i1 * TTOK + p1] = tok; ewt[i1 * TTOK + p1] = v1 / wn;
  }
}

__global__ __launch_bounds__(256) void aux_kernel(
    const float* __restrict__ probs, const int* __restrict__ counts,
    float* __restrict__ out_aux) {
  __shared__ float scratch[8];
  __shared__ float meanp[NEXP];
  int tid = threadIdx.x;
  for (int e = 0; e < NEXP; e++) {
    float s = 0.f;
    for (int t = tid; t < TTOK; t += 256) s += probs[t * NEXP + e];
    s = block_reduce_sum256(s, scratch);
    if (tid == 0) meanp[e] = s / (float)TTOK;
  }
  __syncthreads();
  if (tid == 0) {
    float aux = 0.f;
    for (int e = 0; e < NEXP; e++)
      aux += ((float)counts[e] / (float)(TTOK * 2)) * meanp[e];
    out_aux[0] = (float)NEXP * aux;
  }
}

// ---------------------------------------------------------------------------
// MoE per-expert kernels (fixed grids, device-side early exit on count).
// ---------------------------------------------------------------------------
__global__ __launch_bounds__(256) void moe_gather_kernel(
    const bf16* __restrict__ tbf, const int* __restrict__ counts,
    const int* __restrict__ etok, bf16* __restrict__ Xg, int e) {
  int row = blockIdx.x;
  if (row >= counts[e]) return;
  int tok = etok[e * TTOK + row];
  for (int d = threadIdx.x; d < DMODEL; d += 256)
    Xg[(size_t)row * DMODEL + d] = tbf[(size_t)tok * DMODEL + d];
}

// fused gate+up: ACT = silu(Xg@Wg) * (Xg@Wu), bf16 out. K=1024, N=1408.
__global__ __launch_bounds__(256) void gemm_gateup_kernel(
    const bf16* __restrict__ A, const float* __restrict__ Bg,
    const float* __restrict__ Bu, bf16* __restrict__ ACT,
    const int* __restrict__ counts, int e) {
  const int K = DMODEL, N = INTER;
  int ne = counts[e];
  int m0 = blockIdx.x * 64;
  if (m0 >= ne) return;
  int n0 = blockIdx.y * 128;
  __shared__ __align__(16) bf16 As[64 * 32];
  __shared__ __align__(16) bf16 Btg[128 * 40];
  __shared__ __align__(16) bf16 Btu[128 * 40];
  int tid = threadIdx.x;
  int wid = tid >> 5, lid = tid & 31, l16 = lid & 15, hf = lid >> 4;
  int wm = wid & 3, wn = wid >> 2;
  v8f ag[4] = {}, au[4] = {};
  for (int k0 = 0; k0 < K; k0 += 32) {
    stage_a_tile(As, A, m0, k0, K, tid);
    stage_b_tile(Btg, Bg, k0, n0, N, tid);
    stage_b_tile(Btu, Bu, k0, n0, N, tid);
    stage_a_wait(tid);
    __syncthreads();
    const bf16* ar = &As[(wm * 16 + l16) * 32];
    v16bf afrag = make_frag(ar + 8 * hf, ar + 16 + 8 * hf);
    #pragma unroll
    for (int nt = 0; nt < 4; nt++) {
      const bf16* bg = &Btg[(wn * 64 + nt * 16 + l16) * 40];
      const bf16* bu = &Btu[(wn * 64 + nt * 16 + l16) * 40];
      ag[nt] = wmma_bf16(afrag, make_frag(bg + 8 * hf, bg + 16 + 8 * hf), ag[nt]);
      au[nt] = wmma_bf16(afrag, make_frag(bu + 8 * hf, bu + 16 + 8 * hf), au[nt]);
    }
    __syncthreads();
  }
  #pragma unroll
  for (int nt = 0; nt < 4; nt++)
    #pragma unroll
    for (int r = 0; r < 8; r++) {
      int row = m0 + wm * 16 + r + 8 * hf;
      int col = n0 + wn * 64 + nt * 16 + l16;
      float gv = ag[nt][r], uv = au[nt][r];
      float sv = gv / (1.f + __expf(-gv));      // silu
      ACT[(size_t)row * N + col] = (bf16)(sv * uv);
    }
}

// down projection + weighted scatter-add into output. K=1408, N=1024.
__global__ __launch_bounds__(256) void gemm_down_kernel(
    const bf16* __restrict__ A, const float* __restrict__ Bd,
    float* __restrict__ OUT, const int* __restrict__ counts,
    const int* __restrict__ etok, const float* __restrict__ ewt, int e) {
  const int K = INTER, N = DMODEL;
  int ne = counts[e];
  int m0 = blockIdx.x * 64;
  if (m0 >= ne) return;
  int n0 = blockIdx.y * 128;
  __shared__ __align__(16) bf16 As[64 * 32];
  __shared__ __align__(16) bf16 Bt[128 * 40];
  int tid = threadIdx.x;
  int wid = tid >> 5, lid = tid & 31, l16 = lid & 15, hf = lid >> 4;
  int wm = wid & 3, wn = wid >> 2;
  v8f acc[4] = {};
  for (int k0 = 0; k0 < K; k0 += 32) {
    stage_a_tile(As, A, m0, k0, K, tid);
    stage_b_tile(Bt, Bd, k0, n0, N, tid);
    stage_a_wait(tid);
    __syncthreads();
    const bf16* ar = &As[(wm * 16 + l16) * 32];
    v16bf afrag = make_frag(ar + 8 * hf, ar + 16 + 8 * hf);
    #pragma unroll
    for (int nt = 0; nt < 4; nt++) {
      const bf16* br = &Bt[(wn * 64 + nt * 16 + l16) * 40];
      acc[nt] = wmma_bf16(afrag, make_frag(br + 8 * hf, br + 16 + 8 * hf), acc[nt]);
    }
    __syncthreads();
  }
  #pragma unroll
  for (int nt = 0; nt < 4; nt++)
    #pragma unroll
    for (int r = 0; r < 8; r++) {
      int row = m0 + wm * 16 + r + 8 * hf;
      if (row < ne) {
        int tok = etok[e * TTOK + row];
        float wt = ewt[e * TTOK + row];
        int col = n0 + wn * 64 + nt * 16 + l16;
        atomicAdd(&OUT[(size_t)tok * N + col], wt * acc[nt][r]);
      }
    }
}

// ---------------------------------------------------------------------------
// Host-side orchestration
// ---------------------------------------------------------------------------
extern "C" void kernel_launch(void* const* d_in, const int* in_sizes, int n_in,
                              void* d_out, int out_size, void* d_ws, size_t ws_size,
                              hipStream_t stream) {
  const float* x     = (const float*)d_in[0];
  const float* cosb  = (const float*)d_in[1];
  const float* sinb  = (const float*)d_in[2];
  const float* ln1w  = (const float*)d_in[3];
  const float* wq    = (const float*)d_in[4];
  const float* wk    = (const float*)d_in[5];
  const float* wv    = (const float*)d_in[6];
  const float* wo    = (const float*)d_in[7];
  const float* qnw   = (const float*)d_in[8];
  const float* knw   = (const float*)d_in[9];
  const float* ln2w  = (const float*)d_in[10];
  const float* rw    = (const float*)d_in[11];
  const float* wg    = (const float*)d_in[12];
  const float* wu    = (const float*)d_in[13];
  const float* wd    = (const float*)d_in[14];
  float* out = (float*)d_out;

  char* ws = (char*)d_ws;
  size_t off = 0;
  auto alloc = [&](size_t bytes) -> void* {
    void* p = ws + off;
    off += (bytes + 255) & ~(size_t)255;
    return p;
  };
  bf16*  h_bf   = (bf16*)alloc((size_t)TTOK * DMODEL * 2);
  float* qraw   = (float*)alloc((size_t)TTOK * NHEAD * HD * 4);
  float* kraw   = (float*)alloc((size_t)TTOK * NKV * HD * 4);
  float* vraw   = (float*)alloc((size_t)TTOK * NKV * HD * 4);
  bf16*  qbf    = (bf16*)alloc((size_t)TTOK * NHEAD * HD * 2);
  bf16*  kbf    = (bf16*)alloc((size_t)TTOK * NKV * HD * 2);
  bf16*  vbf    = (bf16*)alloc((size_t)TTOK * NKV * HD * 2);
  bf16*  attnbf = (bf16*)alloc((size_t)TTOK * DMODEL * 2);
  float* woout  = (float*)alloc((size_t)TTOK * DMODEL * 4);
  float* tf32   = (float*)alloc((size_t)TTOK * DMODEL * 4);
  bf16*  tbf    = (bf16*)alloc((size_t)TTOK * DMODEL * 2);
  float* probs  = (float*)alloc((size_t)TTOK * NEXP * 4);
  int*   counts = (int*)alloc(NEXP * 4);
  int*   etok   = (int*)alloc((size_t)NEXP * TTOK * 4);
  float* ewt    = (float*)alloc((size_t)NEXP * TTOK * 4);
  bf16*  Xg     = (bf16*)alloc((size_t)TTOK * DMODEL * 2);
  bf16*  act    = (bf16*)alloc((size_t)TTOK * INTER * 2);
  (void)in_sizes; (void)n_in; (void)out_size; (void)ws_size;

  zero_counts_kernel<<<1, 32, 0, stream>>>(counts);

  // 1) h = rmsnorm(x) -> bf16
  rmsnorm1_kernel<<<TTOK, 256, 0, stream>>>(x, ln1w, h_bf);

  // 2) QKV projections
  gemm_bf16_kernel<<<dim3(TTOK / 64, (NHEAD * HD) / 128), 256, 0, stream>>>(
      h_bf, wq, qraw, TTOK, NHEAD * HD, DMODEL);
  gemm_bf16_kernel<<<dim3(TTOK / 64, (NKV * HD) / 128), 256, 0, stream>>>(
      h_bf, wk, kraw, TTOK, NKV * HD, DMODEL);
  gemm_bf16_kernel<<<dim3(TTOK / 64, (NKV * HD) / 128), 256, 0, stream>>>(
      h_bf, wv, vraw, TTOK, NKV * HD, DMODEL);

  // 3) q/k norm + rope, layout change
  qk_rope_kernel<<<TTOK * 24, 64, 0, stream>>>(
      qraw, kraw, vraw, cosb, sinb, qnw, knw, qbf, kbf, vbf);

  // 4) causal GQA flash attention
  attn_kernel<<<dim3(2 * NHEAD, 1024 / 64), 128, 0, stream>>>(qbf, kbf, vbf, attnbf);

  // 5) output projection + residual
  gemm_bf16_kernel<<<dim3(TTOK / 64, DMODEL / 128), 256, 0, stream>>>(
      attnbf, wo, woout, TTOK, DMODEL, DMODEL);
  add_residual_kernel<<<(TTOK * DMODEL) / 256, 256, 0, stream>>>(
      x, woout, out, TTOK * DMODEL);

  // 6) rmsnorm2 -> t (f32 + bf16)
  rmsnorm2_kernel<<<TTOK, 256, 0, stream>>>(out, ln2w, tf32, tbf);

  // 7) router + aux
  router_kernel<<<TTOK, 256, 0, stream>>>(tf32, rw, probs, counts, etok, ewt);
  aux_kernel<<<1, 256, 0, stream>>>(probs, counts, out + (size_t)TTOK * DMODEL);

  // 8) MoE: per-expert gather -> gate/up (fused SiLU) -> down + scatter-add
  for (int e = 0; e < NEXP; e++) {
    moe_gather_kernel<<<TTOK, 256, 0, stream>>>(tbf, counts, etok, Xg, e);
    gemm_gateup_kernel<<<dim3(TTOK / 64, INTER / 128), 256, 0, stream>>>(
        Xg, wg + (size_t)e * DMODEL * INTER, wu + (size_t)e * DMODEL * INTER,
        act, counts, e);
    gemm_down_kernel<<<dim3(TTOK / 64, DMODEL / 128), 256, 0, stream>>>(
        act, wd + (size_t)e * INTER * DMODEL, out, counts, etok, ewt, e);
  }
}
